// HTPCWM_87454124081375
// MI455X (gfx1250) — matrix operand order repaired and checked by reading, not compiled, not run on hardware
//
#include <hip/hip_runtime.h>
#include <math.h>

// ---------------- config ----------------
#define BATCH 64
#define WPB   8          // waves per block (8 * wave32 = 256 threads)

// d_out offsets (floats)
#define O_FP   0
#define O_L1P  786432
#define O_L2P  1835008
#define O_EF   2359296
#define O_E1   3145728
#define O_E2   4194304
#define O_SC   4718592
#define O_AE   4718596
#define O_IT   4718608

// element counts
#define N_FRAME_PAD 2097152   // B*64*64*8   (NHWC, C padded 3->8)
#define N_L1        1048576   // B*16*16*64
#define N_L2        524288    // B*8*8*128

// means (real element counts)
#define NF_REAL 786432.0f
#define N1_REAL 1048576.0f
#define N2_REAL 524288.0f

// ---- ws layout: fp32 region (float offsets, 64-aligned) ----
#define OFF_FNEXT  0u
#define OFF_FP     2097152u
#define OFF_EF     4194304u
#define OFF_L1P    6291456u
#define OFF_L1HAT  7340032u
#define OFF_E1     8388608u
#define OFF_BU     9437184u
#define OFF_D1     10485760u
#define OFF_L2P    11534336u
#define OFF_L2HAT  12058624u
#define OFF_E2     12582912u
#define OFF_D2     13107200u
#define OFF_ACCS   13631488u  // 36 step accs + 5 tail accs
#define OFF_STATE  13631552u  // prev_e, converged, iters
#define OFF_HALF   13631616u  // f16 region starts here (float offset)

// ---- f16 region offsets (in halfs, all multiples of 8 -> 16B aligned) ----
#define HO_FCUR  0u
#define HO_EF    2097152u
#define HO_E2    4194304u
#define HO_L1P   4718592u
#define HO_L2P   5767168u
#define HO_L2I   6291456u
#define HO_T2    6815744u
#define HO_U1    7340032u
#define HO_S1    8388608u
#define HO_H1    9437184u
#define HO_H2    11534336u
#define HO_W     12058624u

// packed weight sub-offsets (in halfs, relative to HO_W)
#define WO_ENC   0u
#define WO_DEC   8192u
#define WO_DOWN  17408u
#define WO_UP    91136u
#define WO_TR    164864u
#define WO_P1    312320u
#define WO_P2    349184u
#define WO_LP1A  496640u
#define WO_LP1B  570368u
#define WO_LP2A  644096u
#define WO_LP2B  791552u

typedef __attribute__((ext_vector_type(16))) _Float16 v16h;
typedef __attribute__((ext_vector_type(8)))  _Float16 v8h;
typedef __attribute__((ext_vector_type(8)))  float    v8f;

// ---------------- WMMA implicit-GEMM conv (fully constexpr geometry) ----------------
// Y[b,y,x,co] = bias[co] + sum_{ky,kx,ci} W[co,ci,ky,kx] * X[b,(y*STRIDE+ky-PAD)/UP,(x*STRIDE+kx-PAD)/UP,ci]
// X is f16 NHWC (CINP-padded). Weights are pre-packed in exact A-fragment order:
//   Wf[((mtile*CHUNKS + chunk)*32 + lane)*16 + i]
// All 8 waves of a block share one mtile; its fragments are async-staged to LDS once.
template <int CINP, int KSZ, int STRIDE, int PAD, int UP, int SIN>
__device__ __forceinline__ v8h im2col_run8(const _Float16* __restrict__ X, int k0,
                                           int b, int y, int x)
{
  constexpr int LIM = SIN * UP;
  const int pos = k0 / CINP;            // CINP power of two -> shift
  const int cib = k0 - pos * CINP;
  const int ky  = pos / KSZ;            // constant divisor -> folds on unroll
  const int kx  = pos - ky * KSZ;
  const int syf = y * STRIDE + ky - PAD;
  const int sxf = x * STRIDE + kx - PAD;
  v8h r;
  if ((unsigned)syf < (unsigned)LIM && (unsigned)sxf < (unsigned)LIM) {
    const int sy = syf / UP, sx = sxf / UP;     // UP in {1,2,4} -> shift
    r = *(const v8h*)(X + (((size_t)b * SIN + sy) * SIN + sx) * CINP + cib);
  } else {
#pragma unroll
    for (int i = 0; i < 8; ++i) r[i] = (_Float16)0.f;
  }
  return r;
}

template <int SIN, int SOUT, int CINP, int KSZ, int STRIDE, int PAD, int UP,
          int MTILES, int CSTORE>
__global__ void __launch_bounds__(256) conv_wmma_kernel(
    const _Float16* __restrict__ X, const _Float16* __restrict__ Wf,
    const float* __restrict__ bias,
    float* __restrict__ Yf, _Float16* __restrict__ Yh,
    int Cbias, int relu)
{
  constexpr int KPAD   = KSZ * KSZ * CINP;   // multiple of 32
  constexpr int CHUNKS = KPAD / 32;
  constexpr int SS     = SOUT * SOUT;        // power of two
  __shared__ _Float16 asmem[CHUNKS * 512];   // one mtile's A fragments (<=36KB)

  const int wave  = threadIdx.x >> 5;
  const int lane  = threadIdx.x & 31;
  const int mtile = blockIdx.x % MTILES;             // shared by all 8 waves
  const int ntile = (blockIdx.x / MTILES) * WPB + wave;
  const int grp   = lane >> 4;                       // lane group 0/1
  const int lm    = lane & 15;

  // ---- cooperative async stage: global -> LDS (ASYNCcnt-tracked, no VGPR data) ----
  {
    const char* wbase = (const char*)(Wf + (size_t)mtile * CHUNKS * 512);
    __builtin_prefetch((const void*)wbase, 0, 1);    // global_prefetch_b8
    unsigned lbase = (unsigned)(size_t)(void*)asmem; // LDS byte offset (low 32 bits)
    for (int i = threadIdx.x; i < CHUNKS * 64; i += WPB * 32) {
      unsigned    loff = lbase + i * 16;
      const void* g    = wbase + (size_t)i * 16;
      asm volatile("global_load_async_to_lds_b128 %0, %1, off"
                   :: "v"(loff), "v"(g) : "memory");
    }
    asm volatile("s_wait_asynccnt 0x0" ::: "memory");
    __syncthreads();                                 // all waves see staged weights
  }

  // B/D column: one output pixel per lane column
  const int p   = ntile * 16 + lm;
  const int b   = p / SS;
  const int rem = p - b * SS;
  const int y   = rem / SOUT;
  const int x   = rem - y * SOUT;

  v8f acc = {};
#pragma unroll
  for (int c = 0; c < CHUNKS; ++c) {
    const int kk = c * 32;
    // A fragment from LDS: one 32B ds read per chunk
    v16h Af = *(const v16h*)(asmem + ((size_t)c * 32 + lane) * 16);
    // B (32x16 layout): lane grp g holds K = kk+8*g .. +7 and kk+16+8*g .. +7
    v8h b0 = im2col_run8<CINP, KSZ, STRIDE, PAD, UP, SIN>(X, kk + 8 * grp,      b, y, x);
    v8h b1 = im2col_run8<CINP, KSZ, STRIDE, PAD, UP, SIN>(X, kk + 16 + 8 * grp, b, y, x);
    v16h Bf = __builtin_shufflevector(b0, b1, 0,1,2,3,4,5,6,7,8,9,10,11,12,13,14,15);
    acc = __builtin_amdgcn_wmma_f32_16x16x32_f16(false, Af, false, Bf, (short)0, acc, false, false);
  }

  // D layout: lane col N=lm, rows M = 8*grp + j -> 8 contiguous channels
  const int co0 = mtile * 16 + 8 * grp;
  float vj[8];
#pragma unroll
  for (int j = 0; j < 8; ++j) {
    int co = co0 + j;
    float v = acc[j] + (co < Cbias ? bias[co] : 0.f);
    if (relu) v = fmaxf(v, 0.f);
    vj[j] = v;
  }
  if (Yf) {
    float* yp = Yf + (size_t)p * CSTORE + co0;
    if constexpr (CSTORE % 16 == 0) {
      ((float4*)yp)[0] = make_float4(vj[0], vj[1], vj[2], vj[3]);
      ((float4*)yp)[1] = make_float4(vj[4], vj[5], vj[6], vj[7]);
    } else {
      // decode case (CSTORE==8): only lane-group 0 rows exist in the output buffer
#pragma unroll
      for (int j = 0; j < 8; ++j)
        if (co0 + j < CSTORE) yp[j] = vj[j];
    }
  }
  if (Yh) {     // f16 mirror for conv-consumed activations (always CSTORE%16==0 here)
    v8h hv;
#pragma unroll
    for (int j = 0; j < 8; ++j) hv[j] = (_Float16)vj[j];
    *(v8h*)(Yh + (size_t)p * CSTORE + co0) = hv;
  }
}

// ---------------- weight packing: OIHW fp32 -> A-fragment-ordered f16 ----------------
__global__ void pack_w_kernel(const float* __restrict__ w, _Float16* __restrict__ wp,
                              int Cout, int Cin, int ksq, int Cinp, int Coutp)
{
  int idx = blockIdx.x * blockDim.x + threadIdx.x;
  int Kpad = ksq * Cinp;
  if (idx >= Coutp * Kpad) return;
  int chunks = Kpad / 32;
  int i    = idx & 15;
  int lane = (idx >> 4) & 31;
  int mc   = idx >> 9;               // mtile*chunks + chunk
  int c    = mc % chunks;
  int mt   = mc / chunks;
  int grp = lane >> 4, lm = lane & 15;
  int k  = c * 32 + ((i < 8) ? (8 * grp + i) : (16 + 8 * grp + (i - 8)));  // A-layout K
  int co = mt * 16 + lm;
  int pos = k / Cinp;
  int ci  = k - pos * Cinp;
  float v = 0.f;
  if (co < Cout && ci < Cin) v = w[((size_t)co * Cin + ci) * ksq + pos];
  wp[idx] = (_Float16)v;
}

// ---------------- layout kernels ----------------
__global__ void nchw_to_nhwc_kernel(const float* __restrict__ in, float* __restrict__ out,
                                    int C, int Cpad, int S, int n)
{
  int idx = blockIdx.x * blockDim.x + threadIdx.x;
  if (idx >= n) return;
  int c = idx % Cpad;
  int p = idx / Cpad;
  int ss = S * S;
  int b = p / ss, rem = p - b * ss;
  out[idx] = (c < C) ? in[((size_t)b * C + c) * ss + rem] : 0.f;
}

__global__ void nchw_to_nhwc_h_kernel(const float* __restrict__ in, _Float16* __restrict__ out,
                                      int C, int Cpad, int S, int n)
{
  int idx = blockIdx.x * blockDim.x + threadIdx.x;
  if (idx >= n) return;
  int c = idx % Cpad;
  int p = idx / Cpad;
  int ss = S * S;
  int b = p / ss, rem = p - b * ss;
  out[idx] = (c < C) ? (_Float16)in[((size_t)b * C + c) * ss + rem] : (_Float16)0.f;
}

__global__ void nhwc_to_nchw_kernel(const float* __restrict__ in, float* __restrict__ out,
                                    int C, int Cpad, int S, int n)
{
  int idx = blockIdx.x * blockDim.x + threadIdx.x;
  if (idx >= n) return;
  int ss = S * S;
  int rem = idx % ss;
  int c = (idx / ss) % C;
  int b = idx / (ss * C);
  out[idx] = in[((size_t)b * ss + rem) * Cpad + c];
}

// ---------------- elementwise / reduction ----------------
__global__ void ew_sub_kernel(const float* a, const float* b, float* o, int n) {
  int i = blockIdx.x * blockDim.x + threadIdx.x;
  if (i < n) o[i] = a[i] - b[i];
}
__global__ void ew_sub_dual_kernel(const float* a, const float* b, float* o, _Float16* oh, int n) {
  int i = blockIdx.x * blockDim.x + threadIdx.x;
  if (i >= n) return;
  float v = a[i] - b[i];
  o[i] = v;
  oh[i] = (_Float16)v;
}
__global__ void ew_add_h_kernel(const float* a, const float* b, _Float16* oh, int n) {
  int i = blockIdx.x * blockDim.x + threadIdx.x;
  if (i < n) oh[i] = (_Float16)(a[i] + b[i]);
}
__global__ void ew_copy_h_kernel(const _Float16* a, _Float16* o, int n) {
  int i = blockIdx.x * blockDim.x + threadIdx.x;
  if (i < n) o[i] = a[i];
}
// l -= ALPHA * clip(d,-1,1) if not converged; keep f16 mirror in sync
__global__ void ew_update_dual_kernel(float* l, _Float16* lh, const float* d,
                                      const float* st, int n) {
  int i = blockIdx.x * blockDim.x + threadIdx.x;
  if (i >= n) return;
  float g  = 1.f - st[1];
  float dv = fminf(1.f, fmaxf(-1.f, d[i]));
  float v  = l[i] - 0.1f * g * dv;
  l[i]  = v;
  lh[i] = (_Float16)v;
}

__global__ void sumsq_kernel(const float* __restrict__ x, int n, float* acc) {
  __shared__ float sm[256];
  float s = 0.f;
  for (int i = blockIdx.x * blockDim.x + threadIdx.x; i < n; i += gridDim.x * blockDim.x)
    s += x[i] * x[i];
  sm[threadIdx.x] = s;
  __syncthreads();
  for (int o = 128; o > 0; o >>= 1) {
    if (threadIdx.x < o) sm[threadIdx.x] += sm[threadIdx.x + o];
    __syncthreads();
  }
  if (threadIdx.x == 0) atomicAdd(acc, sm[0]);
}

__global__ void init_kernel(float* accs, float* st) {
  int i = threadIdx.x;
  if (i < 41) accs[i] = 0.f;
  if (i == 0) { st[0] = 1e30f; st[1] = 0.f; st[2] = 0.f; }
}

// per-step energy + convergence carry update (scan semantics)
__global__ void finalize_kernel(float* st, const float* acc, float* all_err, int t) {
  if (threadIdx.x != 0 || blockIdx.x != 0) return;
  float e = 1.0f * acc[0] / NF_REAL + 0.1f * (acc[1] / N1_REAL + acc[2] / N2_REAL);
  all_err[t] = e;
  float prev = st[0], conv = st[1], iters = st[2];
  float active = 1.f - conv;
  float rel = fabsf(e - prev) / (prev + 1e-8f);
  float newly = ((rel < 1e-3f) && (iters + 1.f >= 3.f)) ? 1.f : 0.f;
  st[1] = fminf(1.f, conv + newly);
  st[2] = iters + active;
  st[0] = (active > 0.5f) ? e : prev;
}

__global__ void tail_kernel(float* out, const float* tacc, const float* st) {
  if (threadIdx.x != 0 || blockIdx.x != 0) return;
  float ef  = 1.0f * tacc[0] / NF_REAL;
  float e1  = 0.1f * tacc[1] / N1_REAL;
  float e2  = 0.1f * tacc[2] / N2_REAL;
  float reg = 1e-4f * (tacc[3] / N1_REAL + tacc[4] / N2_REAL);
  out[O_SC + 0] = ef + e1 + e2 + reg;
  out[O_SC + 1] = ef;
  out[O_SC + 2] = e1;
  out[O_SC + 3] = e2;
  ((int*)out)[O_IT] = (int)(st[2] + 0.5f);
}

// ---------------- host orchestration ----------------
template <int SIN, int SOUT, int CINP, int KSZ, int STRIDE, int PAD, int UP,
          int MTILES, int CSTORE>
static inline void launch_conv(const _Float16* X, const _Float16* Wf, const void* bias,
                               float* Yf, _Float16* Yh, int Cbias, int relu,
                               hipStream_t stream)
{
  constexpr int tiles = MTILES * (BATCH * SOUT * SOUT / 16);
  conv_wmma_kernel<SIN, SOUT, CINP, KSZ, STRIDE, PAD, UP, MTILES, CSTORE>
      <<<tiles / WPB, 32 * WPB, 0, stream>>>(X, Wf, (const float*)bias, Yf, Yh, Cbias, relu);
}

extern "C" void kernel_launch(void* const* d_in, const int* in_sizes, int n_in,
                              void* d_out, int out_size, void* d_ws, size_t ws_size,
                              hipStream_t stream)
{
  (void)in_sizes; (void)n_in; (void)out_size; (void)ws_size;
  float* ws  = (float*)d_ws;
  float* out = (float*)d_out;

  const float* fcur_in  = (const float*)d_in[0];
  const float* fnext_in = (const float*)d_in[1];

  // fp32 buffers
  float* FNEXT = ws + OFF_FNEXT; float* FP    = ws + OFF_FP;
  float* EF    = ws + OFF_EF;    float* L1P   = ws + OFF_L1P;
  float* L1HAT = ws + OFF_L1HAT; float* E1    = ws + OFF_E1;
  float* BU    = ws + OFF_BU;    float* D1    = ws + OFF_D1;
  float* L2P   = ws + OFF_L2P;   float* L2HAT = ws + OFF_L2HAT;
  float* E2    = ws + OFF_E2;    float* D2    = ws + OFF_D2;
  float* ACC   = ws + OFF_ACCS;          // 36 step accumulators
  float* TACC  = ws + OFF_ACCS + 36;     // 5 tail accumulators
  float* ST    = ws + OFF_STATE;
  // f16 buffers
  _Float16* HB    = (_Float16*)(ws + OFF_HALF);
  _Float16* FCURh = HB + HO_FCUR; _Float16* EFh = HB + HO_EF;
  _Float16* E2h   = HB + HO_E2;   _Float16* L1Ph = HB + HO_L1P;
  _Float16* L2Ph  = HB + HO_L2P;  _Float16* L2Ih = HB + HO_L2I;
  _Float16* T2h   = HB + HO_T2;   _Float16* U1h  = HB + HO_U1;
  _Float16* S1h   = HB + HO_S1;   _Float16* H1h  = HB + HO_H1;
  _Float16* H2h   = HB + HO_H2;   _Float16* WH   = HB + HO_W;

  auto pack = [&](const void* w, unsigned off, int Cout, int Cin, int ksq, int Cinp, int Coutp) {
    int total = Coutp * ksq * Cinp;
    pack_w_kernel<<<(total + 255) / 256, 256, 0, stream>>>(
        (const float*)w, WH + off, Cout, Cin, ksq, Cinp, Coutp);
  };
  auto sumsq = [&](const float* x, int n, float* acc) {
    sumsq_kernel<<<512, 256, 0, stream>>>(x, n, acc);
  };

  // ---- setup ----
  init_kernel<<<1, 64, 0, stream>>>(ACC, ST);
  nchw_to_nhwc_h_kernel<<<(N_FRAME_PAD + 255) / 256, 256, 0, stream>>>(fcur_in,  FCURh, 3, 8, 64, N_FRAME_PAD);
  nchw_to_nhwc_kernel  <<<(N_FRAME_PAD + 255) / 256, 256, 0, stream>>>(fnext_in, FNEXT, 3, 8, 64, N_FRAME_PAD);

  pack(d_in[2],  WO_ENC,  64, 3, 16, 8, 64);     // enc 4x4, Cin 3->8
  pack(d_in[4],  WO_DEC,  3, 64, 9, 64, 16);     // dec, Cout 3->16
  pack(d_in[6],  WO_DOWN, 128, 64, 9, 64, 128);
  pack(d_in[8],  WO_UP,   64, 128, 9, 128, 64);
  pack(d_in[10], WO_TR,   128, 128, 9, 128, 128);
  pack(d_in[12], WO_P1,   64, 64, 9, 64, 64);
  pack(d_in[14], WO_P2,   128, 128, 9, 128, 128);
  pack(d_in[16], WO_LP1A, 128, 64, 9, 64, 128);
  pack(d_in[18], WO_LP1B, 64, 128, 9, 128, 64);
  pack(d_in[20], WO_LP2A, 128, 128, 9, 128, 128);
  pack(d_in[22], WO_LP2B, 128, 128, 9, 128, 128);

  // latent1 = encode(frame_current); latent2 = downsample(latent1); keep init copy (f16)
  launch_conv<64,16,8,4,4,0,1, 4,64>(FCURh, WH + WO_ENC, d_in[3], L1P, L1Ph, 64, 0, stream);
  launch_conv<16,8,64,3,2,0,1, 8,128>(L1Ph, WH + WO_DOWN, d_in[7], L2P, L2Ph, 128, 0, stream);
  ew_copy_h_kernel<<<(N_L2 + 255) / 256, 256, 0, stream>>>(L2Ph, L2Ih, N_L2);

  // ---- 12-step predictive-coding scan ----
  for (int t = 0; t < 12; ++t) {
    launch_conv<16,64,64,3,1,1,4, 1,8>(L1Ph, WH + WO_DEC, d_in[5], FP, (_Float16*)nullptr, 3, 0, stream); // decode
    ew_sub_dual_kernel<<<(N_FRAME_PAD + 255) / 256, 256, 0, stream>>>(FNEXT, FP, EF, EFh, N_FRAME_PAD);   // ef
    launch_conv<8,8,128,3,1,1,1, 8,128>(L2Ph, WH + WO_TR, d_in[11], (float*)nullptr, T2h, 128, 0, stream); // transition
    launch_conv<8,8,128,3,1,1,1, 8,128>(T2h, WH + WO_P2, d_in[15], L2HAT, (_Float16*)nullptr, 128, 0, stream); // pred2
    ew_sub_dual_kernel<<<(N_L2 + 255) / 256, 256, 0, stream>>>(L2P, L2HAT, E2, E2h, N_L2);                 // e2
    launch_conv<8,16,128,3,1,1,2, 4,64>(L2Ph, WH + WO_UP, d_in[9], (float*)nullptr, U1h, 64, 0, stream);   // up
    launch_conv<16,16,64,3,1,1,1, 4,64>(U1h, WH + WO_P1, d_in[13], L1HAT, (_Float16*)nullptr, 64, 0, stream); // pred1
    ew_sub_kernel<<<(N_L1 + 255) / 256, 256, 0, stream>>>(L1P, L1HAT, E1, N_L1);                           // e1
    launch_conv<64,16,8,4,4,0,1, 4,64>(EFh, WH + WO_ENC, d_in[3], BU, (_Float16*)nullptr, 64, 0, stream);  // bottom_up
    ew_add_h_kernel<<<(N_L1 + 255) / 256, 256, 0, stream>>>(E1, BU, S1h, N_L1);                            // s1
    launch_conv<16,16,64,3,1,1,1, 8,128>(S1h, WH + WO_LP1A, d_in[17], (float*)nullptr, H1h, 128, 1, stream); // learned1 relu
    launch_conv<16,16,128,3,1,1,1, 4,64>(H1h, WH + WO_LP1B, d_in[19], D1, (_Float16*)nullptr, 64, 0, stream);
    launch_conv<8,8,128,3,1,1,1, 8,128>(E2h, WH + WO_LP2A, d_in[21], (float*)nullptr, H2h, 128, 1, stream);  // learned2 relu
    launch_conv<8,8,128,3,1,1,1, 8,128>(H2h, WH + WO_LP2B, d_in[23], D2, (_Float16*)nullptr, 128, 0, stream);
    // gated updates read carry flag BEFORE finalize mutates it (stream order)
    ew_update_dual_kernel<<<(N_L1 + 255) / 256, 256, 0, stream>>>(L1P, L1Ph, D1, ST, N_L1);
    ew_update_dual_kernel<<<(N_L2 + 255) / 256, 256, 0, stream>>>(L2P, L2Ph, D2, ST, N_L2);
    sumsq(EF, N_FRAME_PAD, ACC + 3 * t + 0);
    sumsq(E1, N_L1,        ACC + 3 * t + 1);
    sumsq(E2, N_L2,        ACC + 3 * t + 2);
    finalize_kernel<<<1, 1, 0, stream>>>(ST, ACC + 3 * t, out + O_AE, t);
  }

  // ---- tail: final predictions / errors / energies ----
  launch_conv<16,64,64,3,1,1,4, 1,8>(L1Ph, WH + WO_DEC, d_in[5], FP, (_Float16*)nullptr, 3, 0, stream);
  ew_sub_kernel<<<(N_FRAME_PAD + 255) / 256, 256, 0, stream>>>(FNEXT, FP, EF, N_FRAME_PAD);  // error_frame
  launch_conv<8,8,128,3,1,1,1, 8,128>(L2Ih, WH + WO_TR, d_in[11], (float*)nullptr, T2h, 128, 0, stream); // transition(latent2_init)
  launch_conv<8,8,128,3,1,1,1, 8,128>(T2h, WH + WO_P2, d_in[15], L2HAT, (_Float16*)nullptr, 128, 0, stream);
  ew_sub_kernel<<<(N_L2 + 255) / 256, 256, 0, stream>>>(L2P, L2HAT, E2, N_L2);               // error_latent2
  launch_conv<8,16,128,3,1,1,2, 4,64>(L2Ph, WH + WO_UP, d_in[9], (float*)nullptr, U1h, 64, 0, stream);
  launch_conv<16,16,64,3,1,1,1, 4,64>(U1h, WH + WO_P1, d_in[13], L1HAT, (_Float16*)nullptr, 64, 0, stream);
  ew_sub_kernel<<<(N_L1 + 255) / 256, 256, 0, stream>>>(L1P, L1HAT, E1, N_L1);               // error_latent1
  sumsq(EF,  N_FRAME_PAD, TACC + 0);
  sumsq(E1,  N_L1,        TACC + 1);
  sumsq(E2,  N_L2,        TACC + 2);
  sumsq(L1P, N_L1,        TACC + 3);
  sumsq(L2P, N_L2,        TACC + 4);
  tail_kernel<<<1, 1, 0, stream>>>(out, TACC, ST);

  // ---- outputs: NHWC -> NCHW fp32 ----
  nhwc_to_nchw_kernel<<<(786432 + 255) / 256, 256, 0, stream>>>(FP,  out + O_FP,  3, 8, 64, 786432);
  nhwc_to_nchw_kernel<<<(N_L1 + 255) / 256, 256, 0, stream>>>(L1P, out + O_L1P, 64, 64, 16, N_L1);
  nhwc_to_nchw_kernel<<<(N_L2 + 255) / 256, 256, 0, stream>>>(L2P, out + O_L2P, 128, 128, 8, N_L2);
  nhwc_to_nchw_kernel<<<(786432 + 255) / 256, 256, 0, stream>>>(EF,  out + O_EF,  3, 8, 64, 786432);
  nhwc_to_nchw_kernel<<<(N_L1 + 255) / 256, 256, 0, stream>>>(E1,  out + O_E1,  64, 64, 16, N_L1);
  nhwc_to_nchw_kernel<<<(N_L2 + 255) / 256, 256, 0, stream>>>(E2,  out + O_E2,  128, 128, 8, N_L2);
}
